// HierarchicalRecursiveTRM_9809705304257
// MI455X (gfx1250) — matrix operand order repaired and checked by reading, not compiled, not run on hardware
//
#include <hip/hip_runtime.h>
#include <hip/hip_bf16.h>

// ---------------------------------------------------------------------------
// HierarchicalRecursiveTRM for MI455X (gfx1250), wave32 + WMMA f16->f32.
//
// One persistent workgroup (16 waves) owns the 4096-step serial recurrence.
// All recurrent state lives in LDS; activations are staged in LDS as f16
// pre-swizzled into WMMA A-fragment layout (so A loads are 2x ds_load_b128,
// no per-GEMM converts); f16 weights are packed in B-fragment layout in d_ws
// and stay L2-resident across 4096 reuses.
// ---------------------------------------------------------------------------

typedef __attribute__((ext_vector_type(16))) _Float16 v16h;
typedef __attribute__((ext_vector_type(2)))  _Float16 h2;
typedef __attribute__((ext_vector_type(8)))  float    v8f;

#define Bsz 16
#define Ssz 4096
#define Hsz 256

#define GROW 1032   // f32 GEMM outputs : 16 x 1024 (+pad)
#define HROW 264    // f32 state bufs   : 16 x 256  (+pad)
#define APACK_HALVES (16 * 32 * 16)                 // 16 ktiles (K=512 max)
#define SMEM_F32 (16 * GROW + 4 * 16 * HROW)        // 33408 floats
#define SMEM_BYTES (SMEM_F32 * 4 + APACK_HALVES * 2)

// Packed-weight offsets (in _Float16 elements) inside d_ws
#define OFF_WC 0u         // GRU [W_ih | W_hh] : NT=48, KT=16
#define OFF_LG 393216u    // Lg : NT=16, KT=16
#define OFF_LT 524288u
#define OFF_LU 655360u    // Lu : NT=16, KT=8
#define OFF_HG 720896u
#define OFF_HT 851968u
#define OFF_HU 983040u    // total = 1048576 halves = 2MB workspace

// ---------------------------------------------------------------------------
// Weight packing: B-fragment layout for V_WMMA_F32_16X16X32_F16.
// B 32x16 fragment (ISA 7.12.2): lane l holds column n = l&15;
//   half j (0..15) holds k = ktile*32 + 16*(j>>3) + 8*(l>>4) + (j&7).
// ---------------------------------------------------------------------------
__global__ void pack_w_kernel(const float* __restrict__ src, int N, int K,
                              _Float16* __restrict__ dst, int ktile_off, int KT_total) {
    int idx = blockIdx.x * blockDim.x + threadIdx.x;
    if (idx >= N * K) return;
    int n = idx / K, k = idx % K;
    int ntile = n >> 4, nl = n & 15;
    int ktile = (k >> 5) + ktile_off;
    int kr = k & 31;
    int lane = (((kr >> 3) & 1) << 4) + nl;
    int j = ((kr >> 4) << 3) | (kr & 7);
    size_t o = ((((size_t)ntile * KT_total + ktile) * 32) + lane) * 16 + j;
    dst[o] = (_Float16)src[idx];
}

// ---------------------------------------------------------------------------
// A-fragment staging (f16, fragment-layout in LDS).
// A 16x32 fragment (ISA 7.12.2): lane l (m = l&15) half j holds
//   k = ktile*32 + 16*(j>>3) + 8*(l>>4) + (j&7).
// Inverse map for an even column pair (k2, k2+1), row m:
//   ktile=k2>>5, kr=k2&31, j=((kr>>4)<<3)|(kr&7), lane=((kr>>3)&1)*16+m.
// ---------------------------------------------------------------------------
__device__ __forceinline__ void stage_pair(_Float16* __restrict__ apack, int m, int k2,
                                           float lo, float hi) {
    int ktile = k2 >> 5;
    int kr = k2 & 31;
    int j = ((kr >> 4) << 3) | (kr & 7);
    int lane = (((kr >> 3) & 1) << 4) + m;
    h2 v = {(_Float16)lo, (_Float16)hi};           // v_cvt_pk_f16_f32
    *(h2*)(apack + ((ktile * 32 + lane) << 4) + j) = v;  // ds_store_b32
}

__device__ __forceinline__ v16h load_a_packed(const _Float16* __restrict__ apack,
                                              int ktile, int lane) {
    return *(const v16h*)(apack + ((ktile * 32 + lane) << 4)); // 2x ds_load_b128
}

__device__ __forceinline__ v16h load_b_frag(const _Float16* __restrict__ packed,
                                            int ntile, int ktile, int KT, int lane) {
    const _Float16* p = packed + ((((size_t)ntile * KT + ktile) * 32) + lane) * 16;
    return *(const v16h*)p;                        // 2x global_load_b128
}

// C/D 16x16 f32 (ISA 7.12.2): lane l holds n = l&15; elem i holds m = i+8*(l>>4)
__device__ __forceinline__ void store_c(float* __restrict__ buf, int row_stride,
                                        int ncol, v8f c, int lane) {
    int n = ncol + (lane & 15);
    int mbase = 8 * (lane >> 4);
#pragma unroll
    for (int i = 0; i < 8; ++i) buf[(mbase + i) * row_stride + n] = c[i];
}

__device__ __forceinline__ v8f wmma(v16h a, v16h b, v8f c) {
    return __builtin_amdgcn_wmma_f32_16x16x32_f16(false, a, false, b,
                                                  (short)0, c, false, false);
}

// Single-output GEMM tile over ktiles [k0, k0+kt)
__device__ __forceinline__ v8f gemm_tile(const _Float16* __restrict__ apack,
                                         const _Float16* __restrict__ bw,
                                         int ntile, int k0, int kt, int KT, int lane) {
    v8f acc = {};
#pragma unroll 4
    for (int k = 0; k < kt; ++k) {
        v16h a = load_a_packed(apack, k0 + k, lane);
        v16h b = load_b_frag(bw, ntile, k0 + k, KT, lane);
        acc = wmma(a, b, acc);
    }
    return acc;
}

// Two output tiles sharing the same A fragments (one A load, two WMMAs)
__device__ __forceinline__ void gemm_tile2(const _Float16* __restrict__ apack,
                                           const _Float16* __restrict__ bw0, int nt0,
                                           const _Float16* __restrict__ bw1, int nt1,
                                           int kt, int KT, int lane,
                                           v8f& acc0, v8f& acc1) {
#pragma unroll 4
    for (int k = 0; k < kt; ++k) {
        v16h a  = load_a_packed(apack, k, lane);
        v16h b0 = load_b_frag(bw0, nt0, k, KT, lane);
        acc0 = wmma(a, b0, acc0);
        v16h b1 = load_b_frag(bw1, nt1, k, KT, lane);
        acc1 = wmma(a, b1, acc1);
    }
}

__device__ __forceinline__ float fsigm(float x) { return 1.0f / (1.0f + __expf(-x)); }
__device__ __forceinline__ float ftanh(float x) { return 1.0f - 2.0f / (__expf(2.0f * x) + 1.0f); }

// ---------------------------------------------------------------------------
// One refine cell: prev=obuf, hid; out->obuf, new_hid->newhid. 16 waves.
// ---------------------------------------------------------------------------
__device__ __forceinline__ void refine_cell(
    _Float16* apack, float* gbuf, float* obuf, const float* hid, float* newhid,
    const _Float16* Gw, const float* Gb,
    const _Float16* Tw, const float* Tb,
    const _Float16* Uw, const float* Ub,
    int tid, int wave, int lane) {
    // stage c = [prev | hid] into packed-f16 A layout (pair-wise)
    for (int e = tid; e < Bsz * 128; e += 512) {
        int b = e >> 7, jp = (e & 127) << 1;
        stage_pair(apack, b, jp,       obuf[b * HROW + jp], obuf[b * HROW + jp + 1]);
        stage_pair(apack, b, 256 + jp, hid[b * HROW + jp],  hid[b * HROW + jp + 1]);
    }
    __syncthreads();
    {   // gate + refinement GEMMs share A (K=512, 1 N-tile each per wave)
        v8f ag = {}, at = {};
        gemm_tile2(apack, Gw, wave, Tw, wave, 16, 16, lane, ag, at);
        store_c(gbuf, GROW, wave * 16, ag, lane);
        store_c(gbuf, GROW, 256 + wave * 16, at, lane);
    }
    __syncthreads();
    // out = gate*tanh + (1-gate)*prev ; emit packed staging for U GEMM inline
    for (int e = tid; e < Bsz * 128; e += 512) {
        int b = e >> 7, jp = (e & 127) << 1;
        float o2[2];
#pragma unroll
        for (int u = 0; u < 2; ++u) {
            int j = jp + u;
            float g  = fsigm(gbuf[b * GROW + j] + Gb[j]);
            float rf = ftanh(gbuf[b * GROW + 256 + j] + Tb[j]);
            o2[u] = g * rf + (1.0f - g) * obuf[b * HROW + j];
            obuf[b * HROW + j] = o2[u];
        }
        stage_pair(apack, b, jp, o2[0], o2[1]);
    }
    __syncthreads();
    {   // new_hid GEMM (K=256)
        v8f acc = gemm_tile(apack, Uw, wave, 0, 8, 8, lane);
        store_c(gbuf, GROW, 512 + wave * 16, acc, lane);
    }
    __syncthreads();
    for (int e = tid; e < Bsz * Hsz; e += 512) {
        int b = e >> 8, j = e & 255;
        newhid[b * HROW + j] = ftanh(gbuf[b * GROW + 512 + j] + Ub[j]);
    }
    __syncthreads();
}

// ---------------------------------------------------------------------------
// Persistent scan kernel: GRU + hierarchical refinement, fused over S.
// ---------------------------------------------------------------------------
__global__ void __launch_bounds__(512, 1) trm_scan_kernel(
    const float* __restrict__ x, const float* __restrict__ h0,
    const float* __restrict__ gbi, const float* __restrict__ gbh,
    const float* __restrict__ Lgb, const float* __restrict__ Ltb, const float* __restrict__ Lub,
    const float* __restrict__ Hgb, const float* __restrict__ Htb, const float* __restrict__ Hub,
    const _Float16* __restrict__ wpack, float* __restrict__ out) {
    extern __shared__ float smem[];
    float* gbuf = smem;                       // 16 x GROW (f32)
    float* hbuf = gbuf + 16 * GROW;           // 16 x HROW
    float* obuf = hbuf + 16 * HROW;
    float* zHb  = obuf + 16 * HROW;
    float* zLb  = zHb  + 16 * HROW;
    _Float16* apack = (_Float16*)(smem + SMEM_F32);  // 16 ktiles, 32B-aligned

    const int tid = threadIdx.x;
    const int wave = tid >> 5;
    const int lane = tid & 31;

    const _Float16* Wc = wpack + OFF_WC;
    const _Float16* Lg = wpack + OFF_LG;
    const _Float16* Lt = wpack + OFF_LT;
    const _Float16* Lu = wpack + OFF_LU;
    const _Float16* Hg = wpack + OFF_HG;
    const _Float16* Ht = wpack + OFF_HT;
    const _Float16* Hu = wpack + OFF_HU;

    for (int e = tid; e < Bsz * Hsz; e += 512) {
        int b = e >> 8, j = e & 255;
        hbuf[b * HROW + j] = h0[e];
        zHb[b * HROW + j] = 0.0f;
        zLb[b * HROW + j] = 0.0f;
    }
    __syncthreads();

    for (int t = 0; t < Ssz; ++t) {
        // ---- stage [x_t | h] as packed f16; prefetch x_{t+1} ------------
        for (int e = tid; e < Bsz * 128; e += 512) {
            int b = e >> 7, jp = (e & 127) << 1;
            const float* xp = x + ((size_t)b * Ssz + t) * Hsz + jp;
            stage_pair(apack, b, jp, xp[0], xp[1]);
            stage_pair(apack, b, 256 + jp, hbuf[b * HROW + jp], hbuf[b * HROW + jp + 1]);
            if (t + 1 < Ssz) __builtin_prefetch(xp + Hsz, 0, 3); // global_prefetch_b8
        }
        __syncthreads();
        // ---- GRU projections --------------------------------------------
        {   // r,z rows: [x|h] @ [W_ih;W_hh]^T (K=512, ntiles 0..31, 2/wave)
            v8f a0 = {}, a1 = {};
            gemm_tile2(apack, Wc, 2 * wave, Wc, 2 * wave + 1, 16, 16, lane, a0, a1);
            store_c(gbuf, GROW, (2 * wave) * 16, a0, lane);
            store_c(gbuf, GROW, (2 * wave + 1) * 16, a1, lane);
            // n rows split: xn over x (ktiles 0..7), hn over h (ktiles 8..15)
            int nt = 32 + wave;
            v8f axn = gemm_tile(apack, Wc, nt, 0, 8, 16, lane);
            store_c(gbuf, GROW, 512 + wave * 16, axn, lane);
            v8f ahn = gemm_tile(apack, Wc, nt, 8, 8, 16, lane);
            store_c(gbuf, GROW, 768 + wave * 16, ahn, lane);
        }
        __syncthreads();
        // ---- GRU elementwise --------------------------------------------
        for (int e = tid; e < Bsz * Hsz; e += 512) {
            int b = e >> 8, j = e & 255;
            float r  = fsigm(gbuf[b * GROW + j]       + gbi[j]       + gbh[j]);
            float z  = fsigm(gbuf[b * GROW + 256 + j] + gbi[256 + j] + gbh[256 + j]);
            float nn = ftanh(gbuf[b * GROW + 512 + j] + gbi[512 + j]
                             + r * (gbuf[b * GROW + 768 + j] + gbh[512 + j]));
            float hn = (1.0f - z) * nn + z * hbuf[b * HROW + j];
            hbuf[b * HROW + j] = hn;
            obuf[b * HROW + j] = hn;          // o = base_out[t]
        }
        __syncthreads();
        // ---- hierarchical refinement (2 x (3L + 1H)) --------------------
        for (int hc = 0; hc < 2; ++hc) {
            for (int lc = 0; lc < 3; ++lc)
                refine_cell(apack, gbuf, obuf, zHb, zLb,
                            Lg, Lgb, Lt, Ltb, Lu, Lub, tid, wave, lane);
            refine_cell(apack, gbuf, obuf, zLb, zHb,
                        Hg, Hgb, Ht, Htb, Hu, Hub, tid, wave, lane);
        }
        // ---- emit refined output for timestep t -------------------------
        for (int e = tid; e < Bsz * Hsz; e += 512) {
            int b = e >> 8, j = e & 255;
            out[((size_t)b * Ssz + t) * Hsz + j] = obuf[b * HROW + j];
        }
        __syncthreads();
    }
    // final hidden state (1,B,H) appended after (B,S,H)
    for (int e = tid; e < Bsz * Hsz; e += 512) {
        int b = e >> 8, j = e & 255;
        out[(size_t)Bsz * Ssz * Hsz + e] = hbuf[b * HROW + j];
    }
}

// ---------------------------------------------------------------------------
extern "C" void kernel_launch(void* const* d_in, const int* in_sizes, int n_in,
                              void* d_out, int out_size, void* d_ws, size_t ws_size,
                              hipStream_t stream) {
    const float* x    = (const float*)d_in[0];
    const float* h0   = (const float*)d_in[1];
    const float* w_ih = (const float*)d_in[2];
    const float* w_hh = (const float*)d_in[3];
    const float* b_ih = (const float*)d_in[4];
    const float* b_hh = (const float*)d_in[5];
    const float* Lg_w = (const float*)d_in[6];
    const float* Lg_b = (const float*)d_in[7];
    const float* Lt_w = (const float*)d_in[8];
    const float* Lt_b = (const float*)d_in[9];
    const float* Lu_w = (const float*)d_in[10];
    const float* Lu_b = (const float*)d_in[11];
    const float* Hg_w = (const float*)d_in[12];
    const float* Hg_b = (const float*)d_in[13];
    const float* Ht_w = (const float*)d_in[14];
    const float* Ht_b = (const float*)d_in[15];
    const float* Hu_w = (const float*)d_in[16];
    const float* Hu_b = (const float*)d_in[17];
    float* out = (float*)d_out;
    _Float16* wp = (_Float16*)d_ws;   // needs 2MB of workspace

    auto pack = [&](const float* src, int N, int K, unsigned off, int koff, int KT) {
        int n = N * K;
        pack_w_kernel<<<(n + 255) / 256, 256, 0, stream>>>(src, N, K, wp + off, koff, KT);
    };
    pack(w_ih, 768, 256, OFF_WC, 0, 16);
    pack(w_hh, 768, 256, OFF_WC, 8, 16);
    pack(Lg_w, 256, 512, OFF_LG, 0, 16);
    pack(Lt_w, 256, 512, OFF_LT, 0, 16);
    pack(Lu_w, 256, 256, OFF_LU, 0, 8);
    pack(Hg_w, 256, 512, OFF_HG, 0, 16);
    pack(Ht_w, 256, 512, OFF_HT, 0, 16);
    pack(Hu_w, 256, 256, OFF_HU, 0, 8);

    trm_scan_kernel<<<1, 512, SMEM_BYTES, stream>>>(
        x, h0, b_ih, b_hh, Lg_b, Lt_b, Lu_b, Hg_b, Ht_b, Hu_b, wp, out);
}
